// LSTM_39298950758830
// MI455X (gfx1250) — compile-verified, matrix operand order checked by reading, not run on hardware
//
#include <hip/hip_runtime.h>
#include <stdint.h>

// LSTM forward, MI455X (gfx1250, wave32, WMMA + async-to-LDS pipelines).
// B=128, I=512, T=256, H=1024, G=4H=4096.

typedef __bf16 bf16_t;
typedef __attribute__((ext_vector_type(16))) __bf16 v16bf;
typedef __attribute__((ext_vector_type(8)))  float v8f;

#define Bsz 128
#define Isz 512
#define Tsz 256
#define Hsz 1024
#define Gsz 4096
#define Msz (Tsz * Bsz)   // 32768 rows of the xproj GEMM

__device__ __forceinline__ v8f wmma_bf16(v16bf a, v16bf b, v8f c) {
  // D = A(16x32) * B(32x16) + C(16x16), fp32 accumulate
  return __builtin_amdgcn_wmma_f32_16x16x32_bf16(
      /*neg_a=*/false, a, /*neg_b=*/false, b,
      /*c_mod=*/(short)0, c, /*reuse_a=*/false, /*reuse_b=*/false);
}

__device__ __forceinline__ float sigmoidf_fast(float x) {
  return 1.0f / (1.0f + __expf(-x));
}

// ---- CDNA5 async-to-LDS primitives ---------------------------------------
// Flat addresses of LDS objects carry the LDS offset in the low 32 bits
// (ISA 10.2 aperture rules), so we can hand them to the async instruction.

__device__ __forceinline__ void async_cp16(bf16_t* dst_lds, const bf16_t* src_g) {
#if defined(__gfx1250__)
  asm volatile("global_load_async_to_lds_b128 %0, %1, off"
               :: "v"((unsigned)(uintptr_t)dst_lds),
                  "v"((unsigned long long)(uintptr_t)src_g)
               : "memory");
#else
  *(int4*)dst_lds = *(const int4*)src_g;
#endif
}

__device__ __forceinline__ void async_wait_le16() {
#if defined(__gfx1250__)
  asm volatile("s_wait_asynccnt 0x10" ::: "memory");  // <= 2 chunks in flight
#endif
}

__device__ __forceinline__ void async_wait_zero() {
#if defined(__gfx1250__)
  asm volatile("s_wait_asynccnt 0x0" ::: "memory");
#endif
}

// Cooperative async fill of a contiguous LDS region from contiguous global.
// BYTES must be a multiple of 256 threads * 16 B = 4096.
template <int BYTES>
__device__ __forceinline__ void fill_lds_async(bf16_t* lds, const bf16_t* gsrc) {
  const unsigned tid = threadIdx.x;
#pragma unroll
  for (unsigned ofs = 0; ofs < (unsigned)BYTES; ofs += 256u * 16u) {
    const unsigned o = ofs + tid * 16u;
    async_cp16((bf16_t*)((char*)lds + o), (const bf16_t*)((const char*)gsrc + o));
  }
  async_wait_zero();
  __syncthreads();
}

// ---- wave-private, triple-buffered, async-staged 16xK * Kx(4x16) MAC -----
// stage: wave-private LDS region of 3 slots x 2048 bf16 (4 KB each).
// bpN:   per-lane global source of B operand stream for the 4 sub-tiles
//        (already offset by row and the lane's 16-element K half).
// aL:    per-lane LDS pointer into the block-shared A tile.
// Chunk = K-slab of 32: per gate 32 lanes x 32 B = 1 KB; 8 async b128/chunk.
template <int NCHUNK>
__device__ __forceinline__ void gemm4_async(const bf16_t* aL,
                                            const bf16_t* bp0, const bf16_t* bp1,
                                            const bf16_t* bp2, const bf16_t* bp3,
                                            bf16_t* stage, int lane, v8f* acc) {
  const bf16_t* bp[4] = {bp0, bp1, bp2, bp3};
  // prologue: stage chunks 0,1 into slots 0,1
#pragma unroll
  for (int p = 0; p < 2; ++p) {
    bf16_t* slot = stage + p * 2048;
#pragma unroll
    for (int g = 0; g < 4; ++g) {
      const bf16_t* s = bp[g] + p * 32;
      bf16_t* d = slot + g * 512 + lane * 16;
      async_cp16(d, s);
      async_cp16(d + 8, s + 8);
    }
  }
  for (int k = 0; k < NCHUNK; ++k) {
    // stage chunk k+2 (index wraps; the tail writes land in slots that are
    // never read again, so they are harmless and stay in-bounds)
    int j = k + 2; if (j >= NCHUNK) j -= NCHUNK;
    bf16_t* pslot = stage + ((k + 2) % 3) * 2048;
#pragma unroll
    for (int g = 0; g < 4; ++g) {
      const bf16_t* s = bp[g] + j * 32;
      bf16_t* d = pslot + g * 512 + lane * 16;
      async_cp16(d, s);
      async_cp16(d + 8, s + 8);
    }
    // chunk k complete when <=16 outstanding (in-order completion)
    async_wait_le16();
    const bf16_t* slot = stage + (k % 3) * 2048;
    v16bf a = *(const v16bf*)(aL + k * 32);
#pragma unroll
    for (int g = 0; g < 4; ++g) {
      v16bf b = *(const v16bf*)(slot + g * 512 + lane * 16);
      acc[g] = wmma_bf16(a, b, acc[g]);
    }
  }
}

// ---------------------------------------------------------------- converts

__global__ void cvt_f32_bf16(const float* __restrict__ in,
                             bf16_t* __restrict__ out, int n) {
  int i = blockIdx.x * blockDim.x + threadIdx.x;
  if (i < n) out[i] = (bf16_t)in[i];
}

__global__ void bias_sum(const float* __restrict__ a,
                         const float* __restrict__ b,
                         float* __restrict__ out, int n) {
  int i = blockIdx.x * blockDim.x + threadIdx.x;
  if (i < n) out[i] = a[i] + b[i];
}

__global__ void init_state(bf16_t* __restrict__ h0,
                           float* __restrict__ c0, int n) {
  int i = blockIdx.x * blockDim.x + threadIdx.x;
  if (i < n) { h0[i] = (bf16_t)0.0f; c0[i] = 0.0f; }
}

// x: [B, I, T] f32  ->  xT: [T*B, I] bf16   (LDS-tiled transpose on (i,t))
__global__ void __launch_bounds__(256)
transpose_x(const float* __restrict__ x, bf16_t* __restrict__ xT) {
  __shared__ float tile[32][33];
  const int b  = blockIdx.z;
  const int i0 = blockIdx.y * 32;
  const int t0 = blockIdx.x * 32;
  const int tx = threadIdx.x;   // 0..31
  const int ty = threadIdx.y;   // 0..7
#pragma unroll
  for (int r = 0; r < 32; r += 8) {
    tile[ty + r][tx] = x[((size_t)b * Isz + (i0 + ty + r)) * Tsz + (t0 + tx)];
  }
  __syncthreads();
#pragma unroll
  for (int r = 0; r < 32; r += 8) {
    xT[((size_t)(t0 + ty + r) * Bsz + b) * Isz + (i0 + tx)] =
        (bf16_t)tile[tx][ty + r];
  }
}

// ---------------------------------------------------------------- xproj GEMM
// C[m, n] = sum_k A[m,k] * Wih[n,k] + bsum[n]
// A: [32768, 512] bf16 (xT), Wih: [4096, 512] bf16, C: [32768, 4096] f32.
// Block (8 waves) shares one async-staged 16-row A tile in LDS; each wave
// computes a 16(M) x 64(N) tile with a depth-3 async-to-LDS B pipeline.
__global__ void __launch_bounds__(256)
xproj_gemm(const bf16_t* __restrict__ A, const bf16_t* __restrict__ W,
           const float* __restrict__ bsum, float* __restrict__ C) {
  __shared__ __align__(32) bf16_t Atile[16 * Isz];        // 16 KB
  __shared__ __align__(32) bf16_t Bstage[8 * 3 * 2048];   // 96 KB
  const int wave = threadIdx.x >> 5;
  const int lane = threadIdx.x & 31;
  const int m0 = (blockIdx.x >> 3) * 16;                    // 2048 M-tiles
  const int n0 = (((blockIdx.x & 7) << 3) + wave) * 64;     // 64 N-supertiles

  fill_lds_async<16 * Isz * 2>(Atile, A + (size_t)m0 * Isz);

  const int lmod = lane & 15;
  const int koff = (lane >> 4) << 4;         // 0 or 16

  const bf16_t* aL  = Atile + lmod * Isz + koff;            // LDS reads
  const bf16_t* bp0 = W + (size_t)(n0 +  0 + lmod) * Isz + koff;
  const bf16_t* bp1 = W + (size_t)(n0 + 16 + lmod) * Isz + koff;
  const bf16_t* bp2 = W + (size_t)(n0 + 32 + lmod) * Isz + koff;
  const bf16_t* bp3 = W + (size_t)(n0 + 48 + lmod) * Isz + koff;

  v8f acc[4] = {};
  gemm4_async<Isz / 32>(aL, bp0, bp1, bp2, bp3,
                        Bstage + wave * (3 * 2048), lane, acc);

  // C/D layout: lane holds column n = n0' + lane%16; VGPR r -> row m0 + r + 8*(lane/16)
  const int mbase = m0 + ((lane >> 4) << 3);
#pragma unroll
  for (int j = 0; j < 4; ++j) {
    const int n = n0 + 16 * j + lmod;
    const float bias = bsum[n];
#pragma unroll
    for (int r = 0; r < 8; ++r) {
      C[(size_t)(mbase + r) * Gsz + n] = acc[j][r] + bias;
    }
  }
}

// ---------------------------------------------------------------- LSTM step
// gates[b,g] = xp[b,g] + sum_k h[b,k] * Whh[g,k];  fused pointwise update.
// Block (8 waves) shares one async-staged 16-row h tile (32 KB).
// One wave: 16(M=batch) x 16(N=h index) tile, 4 gate accumulators, with the
// W_hh stream (L2-resident: 8 MB bf16 << 192 MB) on a depth-3 async pipeline.
__global__ void __launch_bounds__(256)
lstm_step(const float* __restrict__ xp_t,      // [128, 4096] (this timestep)
          const bf16_t* __restrict__ Whh,      // [4096, 1024]
          const bf16_t* __restrict__ h_in,     // [128, 1024]
          bf16_t* __restrict__ h_out,          // [128, 1024]
          float* __restrict__ c,               // [128, 1024] in/out
          float* __restrict__ hf) {            // [128, 1024] f32 h
  __shared__ __align__(32) bf16_t Atile[16 * Hsz];        // 32 KB
  __shared__ __align__(32) bf16_t Bstage[8 * 3 * 2048];   // 96 KB
  const int wave = threadIdx.x >> 5;
  const int lane = threadIdx.x & 31;
  const int m0 = (blockIdx.x >> 3) * 16;                    // 8 M-tiles
  const int n0 = ((((blockIdx.x & 7) << 3) + wave)) * 16;   // 64 H-tiles

  fill_lds_async<16 * Hsz * 2>(Atile, h_in + (size_t)m0 * Hsz);

  const int lmod = lane & 15;
  const int koff = (lane >> 4) << 4;
  const int n = n0 + lmod;

  // Prefetch the xp gate rows this lane will need in the epilogue.
  {
    const float* p = xp_t + (size_t)(m0 + ((lane >> 4) << 3)) * Gsz + n;
    __builtin_prefetch(p + 0 * Hsz, 0, 0);
    __builtin_prefetch(p + 1 * Hsz, 0, 0);
    __builtin_prefetch(p + 2 * Hsz, 0, 0);
    __builtin_prefetch(p + 3 * Hsz, 0, 0);
  }

  const bf16_t* aL = Atile + lmod * Hsz + koff;             // LDS reads
  const bf16_t* bp0 = Whh + (size_t)(0 * Hsz + n0 + lmod) * Hsz + koff;
  const bf16_t* bp1 = Whh + (size_t)(1 * Hsz + n0 + lmod) * Hsz + koff;
  const bf16_t* bp2 = Whh + (size_t)(2 * Hsz + n0 + lmod) * Hsz + koff;
  const bf16_t* bp3 = Whh + (size_t)(3 * Hsz + n0 + lmod) * Hsz + koff;

  v8f acc[4] = {};
  gemm4_async<Hsz / 32>(aL, bp0, bp1, bp2, bp3,
                        Bstage + wave * (3 * 2048), lane, acc);

  const int mbase = m0 + ((lane >> 4) << 3);
#pragma unroll
  for (int r = 0; r < 8; ++r) {
    const int m = mbase + r;
    const size_t gb = (size_t)m * Gsz;
    const float xi = acc[0][r] + xp_t[gb + 0 * Hsz + n];
    const float xf = acc[1][r] + xp_t[gb + 1 * Hsz + n];
    const float xg = acc[2][r] + xp_t[gb + 2 * Hsz + n];
    const float xo = acc[3][r] + xp_t[gb + 3 * Hsz + n];
    const float ig = sigmoidf_fast(xi);
    const float fg = sigmoidf_fast(xf);
    const float gg = tanhf(xg);
    const float og = sigmoidf_fast(xo);
    const size_t hidx = (size_t)m * Hsz + n;
    const float cc = fg * c[hidx] + ig * gg;
    c[hidx] = cc;
    const float hh = og * tanhf(cc);
    hf[hidx] = hh;
    h_out[hidx] = (bf16_t)hh;
  }
}

// ---------------------------------------------------------------- launcher

static inline size_t align_up(size_t v, size_t a) { return (v + a - 1) & ~(a - 1); }

extern "C" void kernel_launch(void* const* d_in, const int* in_sizes, int n_in,
                              void* d_out, int out_size, void* d_ws, size_t ws_size,
                              hipStream_t stream) {
  (void)in_sizes; (void)n_in; (void)out_size; (void)ws_size;
  const float* x    = (const float*)d_in[0];   // [B, I, T]
  const float* W_ih = (const float*)d_in[1];   // [4H, I]
  const float* W_hh = (const float*)d_in[2];   // [4H, H]
  const float* b_ih = (const float*)d_in[3];   // [4H]
  const float* b_hh = (const float*)d_in[4];   // [4H]
  float* out = (float*)d_out;                  // [B, H]

  char* ws = (char*)d_ws;
  size_t off = 0;
  bf16_t* xT   = (bf16_t*)(ws + off); off = align_up(off + (size_t)Msz * Isz * 2, 256);
  bf16_t* WihB = (bf16_t*)(ws + off); off = align_up(off + (size_t)Gsz * Isz * 2, 256);
  bf16_t* WhhB = (bf16_t*)(ws + off); off = align_up(off + (size_t)Gsz * Hsz * 2, 256);
  float*  bsum = (float*) (ws + off); off = align_up(off + (size_t)Gsz * 4, 256);
  float*  xp   = (float*) (ws + off); off = align_up(off + (size_t)Msz * Gsz * 4, 256);
  bf16_t* hA   = (bf16_t*)(ws + off); off = align_up(off + (size_t)Bsz * Hsz * 2, 256);
  bf16_t* hB   = (bf16_t*)(ws + off); off = align_up(off + (size_t)Bsz * Hsz * 2, 256);
  float*  cbuf = (float*) (ws + off); off = align_up(off + (size_t)Bsz * Hsz * 4, 256);
  float*  hf   = (float*) (ws + off); off = align_up(off + (size_t)Bsz * Hsz * 4, 256);

  // 1) convert weights + biases to staged formats
  {
    int n = Gsz * Isz;
    cvt_f32_bf16<<<(n + 255) / 256, 256, 0, stream>>>(W_ih, WihB, n);
  }
  {
    int n = Gsz * Hsz;
    cvt_f32_bf16<<<(n + 255) / 256, 256, 0, stream>>>(W_hh, WhhB, n);
  }
  bias_sum<<<(Gsz + 255) / 256, 256, 0, stream>>>(b_ih, b_hh, bsum, Gsz);

  // 2) transpose+convert x: [B,I,T] f32 -> [T*B, I] bf16
  {
    dim3 grid(Tsz / 32, Isz / 32, Bsz);
    dim3 blk(32, 8);
    transpose_x<<<grid, blk, 0, stream>>>(x, xT);
  }

  // 3) zero h0 / c0
  {
    int n = Bsz * Hsz;
    init_state<<<(n + 255) / 256, 256, 0, stream>>>(hA, cbuf, n);
  }

  // 4) big input-projection GEMM: xp = xT @ Wih^T + bsum
  {
    // 2048 M-tiles * 8 N-supertile-groups = 16384 blocks, 8 waves/block
    xproj_gemm<<<16384, 256, 0, stream>>>(xT, WihB, bsum, xp);
  }

  // 5) 256 sequential fused recurrent steps (kernel boundary = global sync)
  for (int t = 0; t < Tsz; ++t) {
    const bf16_t* hin = (t & 1) ? hB : hA;
    bf16_t* hout      = (t & 1) ? hA : hB;
    lstm_step<<<64, 256, 0, stream>>>(xp + (size_t)t * Bsz * Gsz,
                                      WhhB, hin, hout, cbuf, hf);
  }

  // 6) final hidden state -> d_out
  hipMemcpyAsync(out, hf, (size_t)Bsz * Hsz * sizeof(float),
                 hipMemcpyDeviceToDevice, stream);
}
